// mLSTMModule_37005438222576
// MI455X (gfx1250) — matrix operand order repaired and checked by reading, not compiled, not run on hardware
//
#include <hip/hip_runtime.h>
#include <math.h>

// ---------------- model constants ----------------
#define E_DIM   512
#define NHEAD   4
#define DQK     256
#define DV      512
#define DHQK    64
#define DHV     128
#define FFN_DIM 1408
#define CHUNK   64
#define B_SZ    8
#define S_IN    4000
#define S_PAD   4032
#define NCHUNK  (S_PAD / CHUNK)
#define M_ROWS  (B_SZ * S_PAD)
#define CAPV    15.0f
#define EPSV    1e-6f
#define NSPLIT  4                 // dv-column split of the mLSTM scan
#define VCOLS   (DHV / NSPLIT)    // 32

typedef __attribute__((ext_vector_type(16))) __bf16 bf16x16;
typedef __attribute__((ext_vector_type(8)))  __bf16 bf16x8;
typedef __attribute__((ext_vector_type(4)))  __bf16 bf16x4;
typedef __attribute__((ext_vector_type(8)))  float  f32x8;
typedef unsigned int u32x4 __attribute__((ext_vector_type(4)));
typedef int          i32x4 __attribute__((ext_vector_type(4)));
typedef int          i32x8 __attribute__((ext_vector_type(8)));

__device__ __forceinline__ __bf16 f2bf(float x) { return (__bf16)x; }
__device__ __forceinline__ float  bf2f(__bf16 x) { return (float)x; }

__device__ __forceinline__ f32x8 wmma_bf16(bf16x16 a, bf16x16 b, f32x8 c) {
  return __builtin_amdgcn_wmma_f32_16x16x32_bf16(false, a, false, b, (short)0, c,
                                                 false, false);
}

// ================= Tensor Data Mover (async 2D tile -> LDS) =================
#if __has_builtin(__builtin_amdgcn_tensor_load_to_lds)
#define USE_TDM 1
#else
#define USE_TDM 0
#endif

#if USE_TDM
typedef __attribute__((address_space(3))) void lds_void_t;
__device__ __forceinline__ unsigned lds_off_of(void* p) {
  return (unsigned)(unsigned long long)(lds_void_t*)p;
}
// Load a (tile_rows x tile_cols) fp32 tile (row stride = stride_elems) from
// global `g` into LDS at byte offset `lds_off`, packed contiguously.
// D# per CDNA5 ISA ch.8: group0 = {count/type, lds_addr, global_addr},
// group1 = {data_size, dims, tile dims, strides}.  2D: groups 2/3 zero.
__device__ __forceinline__ void tdm_load_2d_f32(unsigned lds_off, const float* g,
                                                int tile_rows, int tile_cols,
                                                int stride_elems) {
  unsigned long long ga = (unsigned long long)(const void*)g;
  u32x4 g0;
  g0[0] = 1u;                                        // count=1 (user descriptor)
  g0[1] = lds_off;                                   // lds_addr [63:32]
  g0[2] = (unsigned)(ga & 0xFFFFFFFFull);            // global_addr low
  g0[3] = (unsigned)((ga >> 32) & 0x01FFFFFFull) | (2u << 30);  // addr hi | type=2
  unsigned td0 = (unsigned)tile_cols;                // tensor_dim0 (elements)
  unsigned td1 = (unsigned)tile_rows;                // tensor_dim1
  unsigned long long st0 = (unsigned long long)stride_elems;  // dim0 stride
  i32x8 g1;
  g1[0] = (int)(2u << 16);                           // data_size=2 -> 4B elements
  g1[1] = (int)((td0 & 0xFFFFu) << 16);              // tensor_dim0 low16
  g1[2] = (int)(((td0 >> 16) & 0xFFFFu) | ((td1 & 0xFFFFu) << 16));
  g1[3] = (int)(((td1 >> 16) & 0xFFFFu) | ((unsigned)tile_cols << 16));  // tile_dim0
  g1[4] = (int)((unsigned)tile_rows & 0xFFFFu);      // tile_dim1 (tile_dim2=0)
  g1[5] = (int)(unsigned)(st0 & 0xFFFFFFFFull);      // tensor_dim0_stride low32
  g1[6] = (int)(unsigned)((st0 >> 32) & 0xFFFFull);  // stride hi16 (dim1 stride=0)
  g1[7] = 0;
  i32x4 z4 = {0, 0, 0, 0};
#if __clang_major__ >= 23
  i32x8 z8 = {0, 0, 0, 0, 0, 0, 0, 0};
  __builtin_amdgcn_tensor_load_to_lds(g0, g1, z4, z4, z8, 0);
#else
  __builtin_amdgcn_tensor_load_to_lds(g0, g1, z4, z4, 0);
#endif
}
#endif  // USE_TDM

// ---- fragment loaders (LDS, row-major storage) ----
__device__ __forceinline__ bf16x16 ld_afrag(const __bf16* base, int ldm, int row0,
                                            int k0, int lane) {
  int half = lane >> 4, r = lane & 15;
  const __bf16* p = base + (row0 + r) * ldm + k0;
  bf16x8 lo = *(const bf16x8*)(p + half * 8);
  bf16x8 hi = *(const bf16x8*)(p + 16 + half * 8);
  return __builtin_shufflevector(lo, hi, 0,1,2,3,4,5,6,7,8,9,10,11,12,13,14,15);
}
__device__ __forceinline__ bf16x16 ld_bfrag(const __bf16* base, int ldn, int col0,
                                            int k0, int lane) {
  int khalf = lane >> 4, n = lane & 15;
  const __bf16* p = base + (col0 + n) * ldn + k0;
  bf16x8 lo = *(const bf16x8*)(p + khalf * 8);
  bf16x8 hi = *(const bf16x8*)(p + 16 + khalf * 8);
  return __builtin_shufflevector(lo, hi, 0,1,2,3,4,5,6,7,8,9,10,11,12,13,14,15);
}
__device__ __forceinline__ bf16x16 ld_bfrag_f32(const float* base, int ldn, int col0,
                                                int k0, int lane) {
  int khalf = lane >> 4, n = lane & 15;
  const float* p = base + (col0 + n) * ldn + k0;
  bf16x16 o;
#pragma unroll
  for (int j = 0; j < 8; ++j) {
    o[j]     = f2bf(p[khalf * 8 + j]);
    o[8 + j] = f2bf(p[16 + khalf * 8 + j]);
  }
  return o;
}
__device__ __forceinline__ bf16x16 ld_afrag_rowscale(const __bf16* base, int ldm,
                                                     int row0, int k0, float scale,
                                                     int lane) {
  int half = lane >> 4, r = lane & 15;
  const __bf16* p = base + (row0 + r) * ldm + k0;
  bf16x16 o;
#pragma unroll
  for (int j = 0; j < 8; ++j) {
    o[j]     = f2bf(bf2f(p[half * 8 + j]) * scale);
    o[8 + j] = f2bf(bf2f(p[16 + half * 8 + j]) * scale);
  }
  return o;
}
__device__ __forceinline__ bf16x16 ld_afrag_kscale(const __bf16* base, int ldm,
                                                   int row0, int k0, const float* kf,
                                                   int lane) {
  int half = lane >> 4, r = lane & 15;
  const __bf16* p = base + (row0 + r) * ldm;
  bf16x16 o;
#pragma unroll
  for (int j = 0; j < 8; ++j) {
    int k1 = k0 + half * 8 + j;
    int k2 = k0 + 16 + half * 8 + j;
    o[j]     = f2bf(bf2f(p[k1]) * kf[k1]);
    o[8 + j] = f2bf(bf2f(p[k2]) * kf[k2]);
  }
  return o;
}

// ---------------- pad kernel ----------------
__global__ void pad_k(const float* __restrict__ x, float* __restrict__ xp) {
  int i = blockIdx.x * 256 + threadIdx.x;
  int m = i / E_DIM, c = i % E_DIM;
  int b = m / S_PAD, s = m % S_PAD;
  xp[i] = (s < S_IN) ? x[((size_t)(b * S_IN + s)) * E_DIM + c] : 0.f;
}

// ---------------- RMSNorm ----------------
__global__ __launch_bounds__(256) void rmsnorm_k(const float* __restrict__ x,
                                                 const float* __restrict__ w,
                                                 float* __restrict__ y) {
  __shared__ float red[256];
  int m = blockIdx.x, t = threadIdx.x;
  const float* xr = x + (size_t)m * E_DIM;
  float a0 = xr[t], a1 = xr[t + 256];
  red[t] = a0 * a0 + a1 * a1;
  __syncthreads();
  for (int st = 128; st > 0; st >>= 1) {
    if (t < st) red[t] += red[t + st];
    __syncthreads();
  }
  float inv = rsqrtf(red[0] / (float)E_DIM + EPSV);
  float* yr = y + (size_t)m * E_DIM;
  yr[t] = a0 * inv * w[t];
  yr[t + 256] = a1 * inv * w[t + 256];
}

// ---------------- final RMSNorm + unpad ----------------
__global__ __launch_bounds__(256) void outnorm_k(const float* __restrict__ x,
                                                 const float* __restrict__ w,
                                                 float* __restrict__ out) {
  int m = blockIdx.x, t = threadIdx.x;
  int b = m / S_PAD, s = m % S_PAD;
  if (s >= S_IN) return;
  __shared__ float red[256];
  const float* xr = x + (size_t)m * E_DIM;
  float a0 = xr[t], a1 = xr[t + 256];
  red[t] = a0 * a0 + a1 * a1;
  __syncthreads();
  for (int st = 128; st > 0; st >>= 1) {
    if (t < st) red[t] += red[t + st];
    __syncthreads();
  }
  float inv = rsqrtf(red[0] / (float)E_DIM + EPSV);
  float* orow = out + ((size_t)(b * S_IN + s)) * E_DIM;
  orow[t] = a0 * inv * w[t];
  orow[t + 256] = a1 * inv * w[t + 256];
}

// ---------------- generic WMMA GEMM: C = A(M,K) * W(N,K)^T [+ res] ----------------
#define BM 128
#define BN 64
#define BK 32
#define LDS_STR 40  // bf16 elems per LDS row (16B-aligned frag reads, conflict pad)

__global__ __launch_bounds__(256) void gemm_bf16_wmma(const float* __restrict__ A,
                                                      const float* __restrict__ W,
                                                      const float* __restrict__ res,
                                                      float* __restrict__ C, int N,
                                                      int K) {
  __shared__ __bf16 As[BM * LDS_STR];
  __shared__ __bf16 Bs[BN * LDS_STR];
#if USE_TDM
  __shared__ float Af32[BM * BK];  // TDM staging
  __shared__ float Bf32[BN * BK];
#endif
  int tid = threadIdx.x, lane = tid & 31, w = tid >> 5;
  int wr = w >> 1, wc = w & 1;
  int brow = blockIdx.y * BM, bcol = blockIdx.x * BN;

  f32x8 acc[2][2];
#pragma unroll
  for (int i = 0; i < 2; ++i)
#pragma unroll
    for (int j = 0; j < 2; ++j) acc[i][j] = (f32x8){0.f,0.f,0.f,0.f,0.f,0.f,0.f,0.f};

  for (int k0 = 0; k0 < K; k0 += BK) {
#if USE_TDM
    if (tid == 0) {  // one TDM issue per tile (EXEC-independent DMA)
      tdm_load_2d_f32(lds_off_of(Af32), A + (size_t)brow * K + k0, BM, BK, K);
      tdm_load_2d_f32(lds_off_of(Bf32), W + (size_t)bcol * K + k0, BN, BK, K);
    }
    __builtin_amdgcn_s_wait_tensorcnt(0);
    __syncthreads();
    // staging fp32 -> padded bf16 tiles
#pragma unroll
    for (int it = 0; it < 4; ++it) {
      int idx = tid + it * 256;
      int r = idx >> 3, c4 = idx & 7;
      float4 v = *(const float4*)&Af32[r * BK + c4 * 4];
      *(bf16x4*)&As[r * LDS_STR + c4 * 4] =
          (bf16x4){f2bf(v.x), f2bf(v.y), f2bf(v.z), f2bf(v.w)};
    }
#pragma unroll
    for (int it = 0; it < 2; ++it) {
      int idx = tid + it * 256;
      int r = idx >> 3, c4 = idx & 7;
      float4 v = *(const float4*)&Bf32[r * BK + c4 * 4];
      *(bf16x4*)&Bs[r * LDS_STR + c4 * 4] =
          (bf16x4){f2bf(v.x), f2bf(v.y), f2bf(v.z), f2bf(v.w)};
    }
#else
#pragma unroll
    for (int it = 0; it < 4; ++it) {  // A tile 128x32 fp32 -> bf16
      int idx = tid + it * 256;
      int r = idx >> 3, c4 = idx & 7;
      float4 v = *(const float4*)(A + (size_t)(brow + r) * K + k0 + c4 * 4);
      *(bf16x4*)&As[r * LDS_STR + c4 * 4] =
          (bf16x4){f2bf(v.x), f2bf(v.y), f2bf(v.z), f2bf(v.w)};
    }
#pragma unroll
    for (int it = 0; it < 2; ++it) {  // B tile 64x32
      int idx = tid + it * 256;
      int r = idx >> 3, c4 = idx & 7;
      float4 v = *(const float4*)(W + (size_t)(bcol + r) * K + k0 + c4 * 4);
      *(bf16x4*)&Bs[r * LDS_STR + c4 * 4] =
          (bf16x4){f2bf(v.x), f2bf(v.y), f2bf(v.z), f2bf(v.w)};
    }
#endif
    __syncthreads();
    bf16x16 bf0 = ld_bfrag(Bs, LDS_STR, wc * 32, 0, lane);
    bf16x16 bf1 = ld_bfrag(Bs, LDS_STR, wc * 32 + 16, 0, lane);
#pragma unroll
    for (int mi = 0; mi < 2; ++mi) {
      bf16x16 af = ld_afrag(As, LDS_STR, wr * 32 + mi * 16, 0, lane);
      acc[mi][0] = wmma_bf16(af, bf0, acc[mi][0]);
      acc[mi][1] = wmma_bf16(af, bf1, acc[mi][1]);
    }
    __syncthreads();
  }
  int rsub = lane >> 4, cl = lane & 15;
  if (res) {
#pragma unroll
    for (int mi = 0; mi < 2; ++mi)
#pragma unroll
      for (int ni = 0; ni < 2; ++ni)
#pragma unroll
        for (int r = 0; r < 8; ++r) {
          int row = brow + wr * 32 + mi * 16 + rsub * 8 + r;
          int col = bcol + wc * 32 + ni * 16 + cl;
          C[(size_t)row * N + col] = acc[mi][ni][r] + res[(size_t)row * N + col];
        }
  } else {
#pragma unroll
    for (int mi = 0; mi < 2; ++mi)
#pragma unroll
      for (int ni = 0; ni < 2; ++ni)
#pragma unroll
        for (int r = 0; r < 8; ++r) {
          int row = brow + wr * 32 + mi * 16 + rsub * 8 + r;
          int col = bcol + wc * 32 + ni * 16 + cl;
          C[(size_t)row * N + col] = acc[mi][ni][r];
        }
  }
}

// ---------------- dual GEMM + SiLU gate: out = silu(A Wg^T) * (A Wu^T) ------------
__global__ __launch_bounds__(256) void gemm_gateup_wmma(const float* __restrict__ A,
                                                        const float* __restrict__ Wg,
                                                        const float* __restrict__ Wu,
                                                        float* __restrict__ C, int N,
                                                        int K) {
  __shared__ __bf16 As[BM * LDS_STR];
  __shared__ __bf16 Bg[BN * LDS_STR];
  __shared__ __bf16 Bu[BN * LDS_STR];
  int tid = threadIdx.x, lane = tid & 31, w = tid >> 5;
  int wr = w >> 1, wc = w & 1;
  int brow = blockIdx.y * BM, bcol = blockIdx.x * BN;

  f32x8 accg[2][2], accu[2][2];
#pragma unroll
  for (int i = 0; i < 2; ++i)
#pragma unroll
    for (int j = 0; j < 2; ++j) {
      accg[i][j] = (f32x8){0.f,0.f,0.f,0.f,0.f,0.f,0.f,0.f};
      accu[i][j] = (f32x8){0.f,0.f,0.f,0.f,0.f,0.f,0.f,0.f};
    }

  for (int k0 = 0; k0 < K; k0 += BK) {
#pragma unroll
    for (int it = 0; it < 4; ++it) {
      int idx = tid + it * 256;
      int r = idx >> 3, c4 = idx & 7;
      float4 v = *(const float4*)(A + (size_t)(brow + r) * K + k0 + c4 * 4);
      *(bf16x4*)&As[r * LDS_STR + c4 * 4] =
          (bf16x4){f2bf(v.x), f2bf(v.y), f2bf(v.z), f2bf(v.w)};
    }
#pragma unroll
    for (int it = 0; it < 2; ++it) {
      int idx = tid + it * 256;
      int r = idx >> 3, c4 = idx & 7;
      float4 vg = *(const float4*)(Wg + (size_t)(bcol + r) * K + k0 + c4 * 4);
      float4 vu = *(const float4*)(Wu + (size_t)(bcol + r) * K + k0 + c4 * 4);
      *(bf16x4*)&Bg[r * LDS_STR + c4 * 4] =
          (bf16x4){f2bf(vg.x), f2bf(vg.y), f2bf(vg.z), f2bf(vg.w)};
      *(bf16x4*)&Bu[r * LDS_STR + c4 * 4] =
          (bf16x4){f2bf(vu.x), f2bf(vu.y), f2bf(vu.z), f2bf(vu.w)};
    }
    __syncthreads();
#pragma unroll
    for (int ni = 0; ni < 2; ++ni) {
      bf16x16 bg = ld_bfrag(Bg, LDS_STR, wc * 32 + ni * 16, 0, lane);
      bf16x16 bu = ld_bfrag(Bu, LDS_STR, wc * 32 + ni * 16, 0, lane);
#pragma unroll
      for (int mi = 0; mi < 2; ++mi) {
        bf16x16 af = ld_afrag(As, LDS_STR, wr * 32 + mi * 16, 0, lane);
        accg[mi][ni] = wmma_bf16(af, bg, accg[mi][ni]);
        accu[mi][ni] = wmma_bf16(af, bu, accu[mi][ni]);
      }
    }
    __syncthreads();
  }
  int rsub = lane >> 4, cl = lane & 15;
#pragma unroll
  for (int mi = 0; mi < 2; ++mi)
#pragma unroll
    for (int ni = 0; ni < 2; ++ni)
#pragma unroll
      for (int r = 0; r < 8; ++r) {
        int row = brow + wr * 32 + mi * 16 + rsub * 8 + r;
        int col = bcol + wc * 32 + ni * 16 + cl;
        float g = accg[mi][ni][r], u = accu[mi][ni][r];
        C[(size_t)row * N + col] = (g / (1.f + expf(-g))) * u;
      }
}

// ---------------- small gate projections i,f (NH=4) with bias + softcap ----------
__global__ void gate_proj_k(const float* __restrict__ x, const float* __restrict__ Wi,
                            const float* __restrict__ bi, const float* __restrict__ Wf,
                            const float* __restrict__ bfv, float* __restrict__ gI,
                            float* __restrict__ gF, int Mtot) {
  int idx = blockIdx.x * 256 + threadIdx.x;
  if (idx >= Mtot * 8) return;
  int m = idx >> 3, r = idx & 7, which = r >> 2, h = r & 3;
  const float* Wrow = (which ? Wf : Wi) + h * E_DIM;
  const float* xr = x + (size_t)m * E_DIM;
  float s = 0.f;
#pragma unroll 4
  for (int kk = 0; kk < E_DIM / 4; ++kk) {
    float4 a = ((const float4*)xr)[kk];
    float4 wv = ((const float4*)Wrow)[kk];
    s += a.x * wv.x + a.y * wv.y + a.z * wv.z + a.w * wv.w;
  }
  s += (which ? bfv[h] : bi[h]);
  s = CAPV * tanhf(s / CAPV);
  (which ? gF : gI)[(size_t)m * NHEAD + h] = s;
}

// ---------------- stabilized chunkwise mLSTM scan ----------------
// grid = B*NH*NSPLIT workgroups; each owns VCOLS columns of the state (independent
// along dv), 8 waves, sequential over 63 chunks.  Gate/score math is recomputed
// per split (cheap) to buy 4x occupancy on the v-side matmuls.
__global__ __launch_bounds__(256) void mlstm_chunk_scan(
    const float* __restrict__ qg, const float* __restrict__ kg,
    const float* __restrict__ vg, const float* __restrict__ gi,
    const float* __restrict__ gf, float* __restrict__ hout) {
  __shared__ float  Ct[VCOLS * 64];    // state^T (vcol, d) fp32
  __shared__ __bf16 vT[VCOLS * 64];    // v^T (vcol, s)
  __shared__ __bf16 qb[64 * 64];       // q*scale (l, d)
  __shared__ __bf16 kb[64 * 64];       // k (s, d)
  __shared__ __bf16 kTb[64 * 64];      // k^T (d, s)
  __shared__ __bf16 Sb[64 * 64];       // gated scores (l, s)
  __shared__ float nst[64], bc[64], icv[64], mcomb[64], qfac[64], kfac[64],
      aval[64], invd[64], sc[4];       // sc: [0]=m [1]=decay [2]=g [3]=m_new

  int tid = threadIdx.x, lane = tid & 31, w = tid >> 5;
  int b = blockIdx.x >> 4, head = (blockIdx.x >> 2) & 3;
  int colbase = (blockIdx.x & 3) * VCOLS;

  for (int i = tid; i < VCOLS * 64; i += 256) Ct[i] = 0.f;
  if (tid < 64) nst[tid] = 0.f;
  if (tid == 0) sc[0] = 0.f;
  __syncthreads();

  for (int c = 0; c < NCHUNK; ++c) {
    size_t rowbase = (size_t)b * S_PAD + c * CHUNK;
    // ---- load chunk tiles ----
#pragma unroll
    for (int it = 0; it < 16; ++it) {
      int idx = tid + it * 256;
      int s = idx >> 6, d = idx & 63;
      float qv = qg[(rowbase + s) * DQK + head * DHQK + d] * 0.125f;  // dqk^-0.5
      qb[s * 64 + d] = f2bf(qv);
      float kv = kg[(rowbase + s) * DQK + head * DHQK + d];
      kb[s * 64 + d] = f2bf(kv);
      kTb[d * 64 + s] = f2bf(kv);
    }
#pragma unroll
    for (int it = 0; it < 8; ++it) {
      int idx = tid + it * 256;
      int s = idx >> 5, col = idx & 31;
      vT[col * 64 + s] = f2bf(vg[(rowbase + s) * DV + head * DHV + colbase + col]);
    }
    if (tid < 64) {
      icv[tid] = gi[(rowbase + tid) * NHEAD + head];
      float fv = gf[(rowbase + tid) * NHEAD + head];
      aval[tid] = (fv > 0.f) ? -log1pf(expf(-fv)) : fv - log1pf(expf(fv));  // logsig
    }
    __syncthreads();
    // ---- serial cumsum + prefix max ----
    if (tid == 0) {
      float run = 0.f, pmax = -INFINITY;
      for (int s = 0; s < CHUNK; ++s) {
        run += aval[s];
        bc[s] = run;
        pmax = fmaxf(pmax, icv[s] - run);
        qfac[s] = pmax;  // temp: prefix max of (i - b)
      }
      sc[2] = run;  // g
    }
    __syncthreads();
    float mprev = sc[0], gtot = sc[2];
    if (tid < 64) {
      float m_intra = bc[tid] + qfac[tid];
      float mc = fmaxf(bc[tid] + mprev, m_intra);
      mcomb[tid] = mc;
      qfac[tid] = expf(bc[tid] + mprev - mc);
      aval[tid] = gtot - bc[tid] + icv[tid];  // a
    }
    __syncthreads();
    if (tid == 0) {
      float amax = -INFINITY;
      for (int s = 0; s < CHUNK; ++s) amax = fmaxf(amax, aval[s]);
      float mn = fmaxf(gtot + mprev, amax);
      sc[3] = mn;
      sc[1] = expf(gtot + mprev - mn);  // decay
    }
    __syncthreads();
    float m_new = sc[3];
    if (tid < 64) kfac[tid] = expf(aval[tid] - m_new);
    __syncthreads();

    // ---- Smat = (q kT) * D : 16 tiles, 2 per wave ----
    for (int t = w * 2; t < w * 2 + 2; ++t) {
      int mt = t >> 2, nt = t & 3;
      f32x8 acc = (f32x8){0.f,0.f,0.f,0.f,0.f,0.f,0.f,0.f};
#pragma unroll
      for (int kk = 0; kk < 2; ++kk) {
        bf16x16 af = ld_afrag(qb, 64, mt * 16, kk * 32, lane);
        bf16x16 bfg = ld_bfrag(kb, 64, nt * 16, kk * 32, lane);
        acc = wmma_bf16(af, bfg, acc);
      }
      int s = nt * 16 + (lane & 15), rsub = lane >> 4;
      float bcs = bc[s], ics = icv[s];
#pragma unroll
      for (int r = 0; r < 8; ++r) {
        int l = mt * 16 + rsub * 8 + r;
        float v = 0.f;
        if (s <= l) v = acc[r] * expf(bc[l] - bcs + ics - mcomb[l]);
        Sb[l * 64 + s] = f2bf(v);
      }
    }
    __syncthreads();
    // ---- denominator ----
    if (tid < 64) {
      float sum = 0.f;
      for (int s = 0; s < 64; ++s) sum += bf2f(Sb[tid * 64 + s]);
      float qn = 0.f;
      for (int d = 0; d < 64; ++d) qn += bf2f(qb[tid * 64 + d]) * nst[d];
      float den = sum + qfac[tid] * qn;
      den = fmaxf(fabsf(den), expf(-mcomb[tid])) + EPSV;
      invd[tid] = 1.f / den;
    }
    __syncthreads();
    // ---- num = S v + q_bar C : one 16x16 tile per wave (64 x VCOLS) ----
    {
      int mt = w >> 1, n0 = (w & 1) * 16;
      f32x8 acc = (f32x8){0.f,0.f,0.f,0.f,0.f,0.f,0.f,0.f};
#pragma unroll
      for (int kk = 0; kk < 2; ++kk) {
        bf16x16 bv = ld_bfrag(vT, 64, n0, kk * 32, lane);
        bf16x16 af = ld_afrag(Sb, 64, mt * 16, kk * 32, lane);
        acc = wmma_bf16(af, bv, acc);
      }
#pragma unroll
      for (int kk = 0; kk < 2; ++kk) {
        bf16x16 bC = ld_bfrag_f32(Ct, 64, n0, kk * 32, lane);
        float rs = qfac[mt * 16 + (lane & 15)];
        bf16x16 af = ld_afrag_rowscale(qb, 64, mt * 16, kk * 32, rs, lane);
        acc = wmma_bf16(af, bC, acc);
      }
      int rsub = lane >> 4, cl = lane & 15;
#pragma unroll
      for (int r = 0; r < 8; ++r) {
        int l = mt * 16 + rsub * 8 + r;
        hout[(rowbase + l) * DV + head * DHV + colbase + n0 + cl] =
            acc[r] * invd[l];
      }
    }
    __syncthreads();  // all reads of Ct / nst / tiles done
    // ---- state update: C^T = decay*C^T + (k_bar^T v)^T : one tile per wave ----
    {
      float decay = sc[1];
      int mt = w >> 1, n0 = (w & 1) * 16;
      int d0 = mt * 16 + (lane >> 4) * 8, cidx = n0 + (lane & 15);
      f32x8 acc;
#pragma unroll
      for (int r = 0; r < 8; ++r) acc[r] = decay * Ct[cidx * 64 + d0 + r];
#pragma unroll
      for (int kk = 0; kk < 2; ++kk) {
        bf16x16 af = ld_afrag_kscale(kTb, 64, mt * 16, kk * 32, kfac, lane);
        bf16x16 bv = ld_bfrag(vT, 64, n0, kk * 32, lane);
        acc = wmma_bf16(af, bv, acc);
      }
#pragma unroll
      for (int r = 0; r < 8; ++r) Ct[cidx * 64 + d0 + r] = acc[r];
    }
    if (tid < 64) {
      float decay = sc[1], s = 0.f;
      for (int si = 0; si < 64; ++si) s += bf2f(kb[si * 64 + tid]) * kfac[si];
      nst[tid] = decay * nst[tid] + s;
    }
    if (tid == 0) sc[0] = sc[3];
    __syncthreads();
  }
}

// ---------------- per-head LayerNorm * sigmoid(og) ----------------
__global__ __launch_bounds__(128) void mhln_gate_k(const float* __restrict__ h,
                                                   const float* __restrict__ og,
                                                   const float* __restrict__ w,
                                                   float* __restrict__ y) {
  __shared__ float red[128];
  int blk = blockIdx.x, t = threadIdx.x;
  int m = blk >> 2, head = blk & 3;
  size_t base = (size_t)m * DV + head * DHV;
  float v = h[base + t];
  red[t] = v;
  __syncthreads();
  for (int st = 64; st > 0; st >>= 1) {
    if (t < st) red[t] += red[t + st];
    __syncthreads();
  }
  float mu = red[0] / (float)DHV;
  __syncthreads();
  float dv = v - mu;
  red[t] = dv * dv;
  __syncthreads();
  for (int st = 64; st > 0; st >>= 1) {
    if (t < st) red[t] += red[t + st];
    __syncthreads();
  }
  float var = red[0] / (float)DHV;
  float yn = dv * rsqrtf(var + EPSV) * w[head * DHV + t];
  float o = og[base + t];
  y[base + t] = yn / (1.f + expf(-o));
}

// ---------------- host launch ----------------
extern "C" void kernel_launch(void* const* d_in, const int* in_sizes, int n_in,
                              void* d_out, int out_size, void* d_ws, size_t ws_size,
                              hipStream_t stream) {
  (void)in_sizes; (void)n_in; (void)out_size; (void)ws_size;
  const float* xin = (const float*)d_in[0];
  const float* out_norm_w = (const float*)d_in[1 + 2 * 15];

  const size_t M = M_ROWS;
  float* ws = (float*)d_ws;
  float* X  = ws;
  float* Y  = X + M * 512;
  float* T0 = Y + M * 512;
  float* T1 = T0 + M * 512;
  float* G  = T1 + M * 512;   // holds q|k|v during attention, act during FFN
  float* gI = G + M * 1408;
  float* gF = gI + M * 4;
  float* Gq = G;
  float* Gk = G + M * 256;
  float* Gv = G + M * 512;

  pad_k<<<(int)(M * 512 / 256), 256, 0, stream>>>(xin, X);

  for (int blk = 0; blk < 2; ++blk) {
    const float* const* P = (const float* const*)(d_in + 1 + blk * 15);
    const float *n1w = P[0], *Wq = P[1], *Wk = P[2], *Wv = P[3], *Wi = P[4],
                *bi = P[5], *Wf = P[6], *bfv = P[7], *mh_w = P[8], *Wog = P[9],
                *Wout = P[10], *n2w = P[11], *Wg = P[12], *Wu = P[13], *Wd = P[14];

    rmsnorm_k<<<(int)M, 256, 0, stream>>>(X, n1w, T0);

    dim3 blkDim(256);
    gemm_bf16_wmma<<<dim3(DQK / BN, M / BM), blkDim, 0, stream>>>(T0, Wq, nullptr, Gq,
                                                                  DQK, 512);
    gemm_bf16_wmma<<<dim3(DQK / BN, M / BM), blkDim, 0, stream>>>(T0, Wk, nullptr, Gk,
                                                                  DQK, 512);
    gemm_bf16_wmma<<<dim3(DV / BN, M / BM), blkDim, 0, stream>>>(T0, Wv, nullptr, Gv,
                                                                 DV, 512);
    gemm_bf16_wmma<<<dim3(DV / BN, M / BM), blkDim, 0, stream>>>(T0, Wog, nullptr, T1,
                                                                 DV, 512);
    gate_proj_k<<<(int)((M * 8 + 255) / 256), 256, 0, stream>>>(T0, Wi, bi, Wf, bfv,
                                                                gI, gF, (int)M);
    mlstm_chunk_scan<<<B_SZ * NHEAD * NSPLIT, 256, 0, stream>>>(Gq, Gk, Gv, gI, gF,
                                                                T0);
    mhln_gate_k<<<(int)(M * NHEAD), 128, 0, stream>>>(T0, T1, mh_w, T1);
    gemm_bf16_wmma<<<dim3(E_DIM / BN, M / BM), blkDim, 0, stream>>>(T1, Wout, X, Y,
                                                                    E_DIM, 512);
    rmsnorm_k<<<(int)M, 256, 0, stream>>>(Y, n2w, T0);
    gemm_gateup_wmma<<<dim3(FFN_DIM / BN, M / BM), blkDim, 0, stream>>>(T0, Wg, Wu, G,
                                                                        FFN_DIM, 512);
    gemm_bf16_wmma<<<dim3(E_DIM / BN, M / BM), blkDim, 0, stream>>>(G, Wd, Y, X,
                                                                    E_DIM, FFN_DIM);
  }
  outnorm_k<<<(int)M, 256, 0, stream>>>(X, out_norm_w, (float*)d_out);
}
